// Transformer_48438641164925
// MI455X (gfx1250) — compile-verified, matrix operand order checked by reading, not run on hardware
//
#include <hip/hip_runtime.h>
#include <hip/hip_fp16.h>
#include <math.h>

// ---------------------------------------------------------------------------
// CDNA5 (gfx1250) wave32 WMMA transformer forward.
// ---------------------------------------------------------------------------

typedef __attribute__((ext_vector_type(16))) _Float16 v16h;
typedef __attribute__((ext_vector_type(8)))  float    v8f;

#define WMMA_F32_F16(a, b, c) \
  __builtin_amdgcn_wmma_f32_16x16x32_f16(false, (a), false, (b), (short)0, (c), false, false)

constexpr int BATCH  = 64;
constexpr int NCAT   = 128;   // NC
constexpr int DMODEL = 256;   // D == INNER
constexpr int NHEAD  = 8;
constexpr int DHEAD  = 32;
constexpr int NTOK   = 256;   // N = 2*NC
constexpr int FFD    = 1024;  // FF
constexpr int NLAYER = 4;
constexpr int BN     = BATCH * NTOK;   // 16384 tokens
constexpr float ATT_SCALE = 0.17677669529663687f;  // DH^-0.5
constexpr float LN_EPS    = 1e-5f;

// --- WMMA fragment loader -------------------------------------------------
// Row-major f16 matrix, 16 rows x 32 cols tile -> v16h per ISA 16-bit A/B layout:
// lane 0-15  : row = lane,    halves 0-7 = K0..7,  halves 8-15 = K16..23
// lane 16-31 : row = lane-16, halves 0-7 = K8..15, halves 8-15 = K24..31
// Each lane: two contiguous 16-byte chunks.
__device__ inline v16h load_frag(const _Float16* p, int ld, int row0, int col0, int lane) {
  int m  = lane & 15;
  int hi = lane >> 4;
  const _Float16* r = p + (size_t)(row0 + m) * (size_t)ld + col0 + hi * 8;
  union { uint4 u[2]; v16h v; } u;
  u.u[0] = *reinterpret_cast<const uint4*>(r);
  u.u[1] = *reinterpret_cast<const uint4*>(r + 16);
  return u.v;
}

// --- concat [B,NC,D] + [B,NC,D] -> [B,2NC,D] fp32 ---------------------------
__global__ __launch_bounds__(256) void concat_kernel(
    const float* __restrict__ x, const float* __restrict__ xc, float* __restrict__ out) {
  size_t i = (size_t)blockIdx.x * blockDim.x + threadIdx.x;
  if (i >= (size_t)BN * DMODEL) return;
  int d = (int)(i % DMODEL);
  size_t t = i / DMODEL;
  int n = (int)(t % NTOK);
  int b = (int)(t / NTOK);
  float v;
  if (n < NCAT) v = x [((size_t)b * NCAT + n)        * DMODEL + d];
  else          v = xc[((size_t)b * NCAT + (n-NCAT)) * DMODEL + d];
  out[i] = v;
}

// --- weight convert + transpose: W[K][Nc] f32 -> WT[Nc][K] f16 -------------
__global__ __launch_bounds__(256) void cvt_t_kernel(
    const float* __restrict__ src, _Float16* __restrict__ dst, int K, int Nc) {
  int i = blockIdx.x * blockDim.x + threadIdx.x;
  if (i >= K * Nc) return;
  int n = i / K, k = i % K;
  dst[i] = (_Float16)src[(size_t)k * Nc + n];
}

// --- LayerNorm: one wave per token (D=256 -> 8 elems/lane) -----------------
__global__ __launch_bounds__(128) void ln_kernel(
    const float* __restrict__ x, const float* __restrict__ g, const float* __restrict__ bt,
    float* __restrict__ hres, _Float16* __restrict__ h16) {
  int lane = threadIdx.x & 31, wave = threadIdx.x >> 5;
  int t = blockIdx.x * 4 + wave;
  const float* xr = x + (size_t)t * DMODEL;
  float v[8];
  float s = 0.f;
  #pragma unroll
  for (int i = 0; i < 8; ++i) { v[i] = xr[lane + i * 32]; s += v[i]; }
  #pragma unroll
  for (int off = 16; off > 0; off >>= 1) s += __shfl_xor(s, off, 32);
  float mean = s * (1.0f / DMODEL);
  float vs = 0.f;
  #pragma unroll
  for (int i = 0; i < 8; ++i) { float d = v[i] - mean; vs += d * d; }
  #pragma unroll
  for (int off = 16; off > 0; off >>= 1) vs += __shfl_xor(vs, off, 32);
  float inv = rsqrtf(vs * (1.0f / DMODEL) + LN_EPS);
  #pragma unroll
  for (int i = 0; i < 8; ++i) {
    int d = lane + i * 32;
    float y = (v[i] - mean) * inv * g[d] + bt[d];
    hres[(size_t)t * DMODEL + d] = y;
    h16 [(size_t)t * DMODEL + d] = (_Float16)y;
  }
}

// --- generic WMMA GEMM: C[M][Nn] = A16[M][K] x BT16[Nn][K] (+bias +res) ----
// 4 waves/block; wave w computes a 32x64 tile: rows [bx*128+32w,+32) x
// cols [by*64,+64). Every B-fragment is reused by two A-fragments -> 8 WMMAs
// per 6 fragment loads per K-chunk.
__global__ __launch_bounds__(128) void gemm16(
    const _Float16* __restrict__ A, int lda,
    const _Float16* __restrict__ BT, int ldb, int K,
    float* __restrict__ outF, _Float16* __restrict__ outH, int ldo,
    const float* __restrict__ bias, const float* __restrict__ res) {
  int lane = threadIdx.x & 31, wave = threadIdx.x >> 5;
  int m0 = blockIdx.x * 128 + wave * 32;
  int n0 = blockIdx.y * 64;
  v8f acc[2][4] = {};
  for (int k0 = 0; k0 < K; k0 += 32) {
    if (k0 + 32 < K) {
      __builtin_prefetch(A  + (size_t)(m0 + (lane & 15)) * lda + k0 + 32, 0, 1);
      __builtin_prefetch(BT + (size_t)(n0 + (lane & 15)) * ldb + k0 + 32, 0, 1);
    }
    v16h a0 = load_frag(A, lda, m0,      k0, lane);
    v16h a1 = load_frag(A, lda, m0 + 16, k0, lane);
    v16h b0 = load_frag(BT, ldb, n0,      k0, lane);
    v16h b1 = load_frag(BT, ldb, n0 + 16, k0, lane);
    v16h b2 = load_frag(BT, ldb, n0 + 32, k0, lane);
    v16h b3 = load_frag(BT, ldb, n0 + 48, k0, lane);
    acc[0][0] = WMMA_F32_F16(a0, b0, acc[0][0]);
    acc[1][0] = WMMA_F32_F16(a1, b0, acc[1][0]);
    acc[0][1] = WMMA_F32_F16(a0, b1, acc[0][1]);
    acc[1][1] = WMMA_F32_F16(a1, b1, acc[1][1]);
    acc[0][2] = WMMA_F32_F16(a0, b2, acc[0][2]);
    acc[1][2] = WMMA_F32_F16(a1, b2, acc[1][2]);
    acc[0][3] = WMMA_F32_F16(a0, b3, acc[0][3]);
    acc[1][3] = WMMA_F32_F16(a1, b3, acc[1][3]);
  }
  int hi = lane >> 4, n = lane & 15;
  #pragma unroll
  for (int i = 0; i < 2; ++i) {
    #pragma unroll
    for (int j = 0; j < 4; ++j) {
      #pragma unroll
      for (int r = 0; r < 8; ++r) {
        int row = m0 + 16 * i + r + 8 * hi;  // C layout: M = r + 8*(lane>=16)
        int col = n0 + 16 * j + n;           //           N = lane & 15
        float v = acc[i][j][r];
        if (bias) v += bias[col];
        if (res)  v += res[(size_t)row * DMODEL + col];
        if (outF) outF[(size_t)row * ldo + col] = v;
        if (outH) outH[(size_t)row * ldo + col] = (_Float16)v;
      }
    }
  }
}

// --- fused FF1 + GEGLU: u = h2 @ Wff1 + b; out = a * gelu_exact(g) (f16) ---
__global__ __launch_bounds__(128) void ff1_geglu_kernel(
    const _Float16* __restrict__ A, const _Float16* __restrict__ BT,
    const float* __restrict__ bias, _Float16* __restrict__ outH) {
  int lane = threadIdx.x & 31, wave = threadIdx.x >> 5;
  int m0 = blockIdx.x * 64 + wave * 16;
  int n0 = blockIdx.y * 64;                 // gate column offset in [0, FFD)
  v8f accA[4] = {};
  v8f accG[4] = {};
  for (int k0 = 0; k0 < DMODEL; k0 += 32) {
    v16h a = load_frag(A, DMODEL, m0, k0, lane);
    #pragma unroll
    for (int j = 0; j < 4; ++j) {
      v16h bA = load_frag(BT, DMODEL, n0 + 16 * j, k0, lane);
      accA[j] = WMMA_F32_F16(a, bA, accA[j]);
      v16h bG = load_frag(BT, DMODEL, FFD + n0 + 16 * j, k0, lane);
      accG[j] = WMMA_F32_F16(a, bG, accG[j]);
    }
  }
  int hi = lane >> 4, n = lane & 15;
  #pragma unroll
  for (int j = 0; j < 4; ++j) {
    #pragma unroll
    for (int r = 0; r < 8; ++r) {
      int row = m0 + r + 8 * hi;
      int col = n0 + 16 * j + n;
      float av = accA[j][r] + bias[col];
      float gv = accG[j][r] + bias[col + FFD];
      float ge = 0.5f * gv * (1.0f + erff(gv * 0.70710678118654752f));
      outH[(size_t)row * FFD + col] = (_Float16)(av * ge);
    }
  }
}

// --- attention: one workgroup per (batch, head); 2 waves -------------------
// smem: [wave0 sim 16KB][wave1 sim 16KB][VT 16.5KB][ips 1KB][keep 1KB][allt]
constexpr int ATT_OFF_VT   = 32768;
constexpr int ATT_OFF_IPS  = ATT_OFF_VT + 32 * 264 * 2;   // 49664
constexpr int ATT_OFF_KEEP = ATT_OFF_IPS + 1024;          // 50688
constexpr int ATT_OFF_ALLT = ATT_OFF_KEEP + 1024;         // 51712
constexpr int ATT_SMEM     = ATT_OFF_ALLT + 32;           // 51744 bytes

__global__ __launch_bounds__(64) void attn_kernel(
    const _Float16* __restrict__ qkv,
    const int* __restrict__ cat_mask, const int* __restrict__ cont_mask,
    const float* __restrict__ cat_ips, const float* __restrict__ con_ips,
    _Float16* __restrict__ o16) {
  extern __shared__ char smem[];
  const int b = blockIdx.x, h = blockIdx.y;
  const int tid = threadIdx.x, lane = tid & 31, wave = tid >> 5;

  float*    ipsS  = (float*)(smem + ATT_OFF_IPS);
  int*      keepS = (int*)  (smem + ATT_OFF_KEEP);
  int*      alltS = (int*)  (smem + ATT_OFF_ALLT);
  _Float16* VT    = (_Float16*)(smem + ATT_OFF_VT);       // [32][264] = V^T
  float*    sim   = (float*)(smem + wave * 16384);        // per-wave [16][256]
  _Float16* attnb = (_Float16*)sim;                       // f16 attn reuses sim

  // masks / ips for this batch row
  for (int j = tid; j < NTOK; j += blockDim.x) {
    int kp; float ip;
    if (j < NCAT) { kp = cat_mask[b * NCAT + j];          ip = cat_ips[b * NCAT + j]; }
    else          { kp = cont_mask[b * NCAT + j - NCAT];  ip = con_ips[b * NCAT + j - NCAT]; }
    keepS[j] = kp; ipsS[j] = ip;
  }
  if (tid == 0) *alltS = 1;
  __syncthreads();
  for (int j = tid; j < NTOK; j += blockDim.x)
    if (!keepS[j]) atomicAnd(alltS, 0);
  // stage V^T into LDS so attn@V B-fragments are contiguous
  for (int i = tid; i < NTOK * DHEAD; i += blockDim.x) {
    int j = i >> 5, d = i & 31;
    VT[d * 264 + j] = qkv[((size_t)(b * NTOK + j)) * 768 + 512 + h * DHEAD + d];
  }
  __syncthreads();
  const int allt = *alltS;

  const _Float16* Q  = qkv + h * DHEAD;          // ld 768, rows b*NTOK+m
  const _Float16* Kp = qkv + 256 + h * DHEAD;    // keys
  const int hi = lane >> 4, nn = lane & 15;

  for (int qb = wave; qb < 16; qb += 2) {
    // ---- scores: sim[16][256] = Q_tile (16x32) x K^T ----
    v16h aq = load_frag(Q, 768, b * NTOK + qb * 16, 0, lane);
    for (int j0 = 0; j0 < NTOK; j0 += 16) {
      v16h bk = load_frag(Kp, 768, b * NTOK + j0, 0, lane);
      v8f c = {};
      c = WMMA_F32_F16(aq, bk, c);
      #pragma unroll
      for (int r = 0; r < 8; ++r)
        sim[(r + 8 * hi) * 256 + j0 + nn] = c[r];
    }
    // ---- masked softmax per row (8 cols/lane, wave32 shfl reductions) ----
    for (int r = 0; r < 16; ++r) {
      float vals[8];
      float mx = -INFINITY;
      #pragma unroll
      for (int i = 0; i < 8; ++i) {
        int j = lane + i * 32;
        float s = sim[r * 256 + j] * ATT_SCALE;
        if (!allt) s = keepS[j] ? s * ipsS[j] : -INFINITY;
        vals[i] = s;
        mx = fmaxf(mx, s);
      }
      #pragma unroll
      for (int off = 16; off > 0; off >>= 1) mx = fmaxf(mx, __shfl_xor(mx, off, 32));
      float sum = 0.f;
      #pragma unroll
      for (int i = 0; i < 8; ++i) { float e = expf(vals[i] - mx); vals[i] = e; sum += e; }
      #pragma unroll
      for (int off = 16; off > 0; off >>= 1) sum += __shfl_xor(sum, off, 32);
      float inv = 1.0f / sum;
      #pragma unroll
      for (int i = 0; i < 8; ++i)
        attnb[r * 256 + lane + i * 32] = (_Float16)(vals[i] * inv);
    }
    // ---- o_tile[16][32] = attn (16x256) x V (256x32) ----
    v8f acc0 = {}, acc1 = {};
    for (int j0 = 0; j0 < NTOK; j0 += 32) {
      v16h af = load_frag(attnb, 256, 0, j0, lane);
      v16h b0 = load_frag(VT, 264, 0,  j0, lane);
      v16h b1 = load_frag(VT, 264, 16, j0, lane);
      acc0 = WMMA_F32_F16(af, b0, acc0);
      acc1 = WMMA_F32_F16(af, b1, acc1);
    }
    #pragma unroll
    for (int r = 0; r < 8; ++r) {
      size_t row = (size_t)(b * NTOK + qb * 16 + r + 8 * hi);
      o16[row * DMODEL + h * DHEAD + nn]      = (_Float16)acc0[r];
      o16[row * DMODEL + h * DHEAD + 16 + nn] = (_Float16)acc1[r];
    }
  }
}

// ---------------------------------------------------------------------------
extern "C" void kernel_launch(void* const* d_in, const int* in_sizes, int n_in,
                              void* d_out, int out_size, void* d_ws, size_t ws_size,
                              hipStream_t stream) {
  (void)in_sizes; (void)n_in; (void)out_size; (void)ws_size;
  const float* x        = (const float*)d_in[0];
  const float* x_cont   = (const float*)d_in[1];
  const int*   cat_mask = (const int*)  d_in[2];
  const int*   con_mask = (const int*)  d_in[3];
  const float* cat_ips  = (const float*)d_in[4];
  const float* con_ips  = (const float*)d_in[5];
  const float* ln1_g    = (const float*)d_in[6];
  const float* ln1_b    = (const float*)d_in[7];
  const float* w_qkv    = (const float*)d_in[8];
  const float* w_out    = (const float*)d_in[9];
  const float* b_out    = (const float*)d_in[10];
  const float* ln2_g    = (const float*)d_in[11];
  const float* ln2_b    = (const float*)d_in[12];
  const float* w_ff1    = (const float*)d_in[13];
  const float* b_ff1    = (const float*)d_in[14];
  const float* w_ff2    = (const float*)d_in[15];
  const float* b_ff2    = (const float*)d_in[16];
  float* out = (float*)d_out;

  // scratch carve (all offsets 256B aligned); total = 92,274,688 bytes
  char* ws = (char*)d_ws;
  float*    X    = (float*)   (ws + 0);          // [BN][D] fp32        16 MB
  float*    HRES = (float*)   (ws + 16777216);   // normed input fp32   16 MB
  _Float16* H16  = (_Float16*)(ws + 33554432);   // normed input f16     8 MB
  _Float16* BIG  = (_Float16*)(ws + 41943040);   // qkv[BN][768]/ffin[BN][1024] 32 MB
  _Float16* O16  = (_Float16*)(ws + 75497472);   // attn out f16         8 MB
  _Float16* WQ   = (_Float16*)(ws + 83886080);   // wqkv^T  [L][768][256]
  _Float16* WO   = (_Float16*)(ws + 85458944);   // wout^T  [L][256][256]
  _Float16* WF1  = (_Float16*)(ws + 85983232);   // wff1^T  [L][2048][256]
  _Float16* WF2  = (_Float16*)(ws + 90177536);   // wff2^T  [L][256][1024]

  // weights -> f16, transposed to [N][K] so B-fragments load contiguously
  for (int l = 0; l < NLAYER; ++l) {
    { int n = DMODEL * 768;
      cvt_t_kernel<<<(n + 255) / 256, 256, 0, stream>>>(
        w_qkv + (size_t)l * DMODEL * 768, WQ + (size_t)l * 768 * DMODEL, DMODEL, 768); }
    { int n = DMODEL * DMODEL;
      cvt_t_kernel<<<(n + 255) / 256, 256, 0, stream>>>(
        w_out + (size_t)l * DMODEL * DMODEL, WO + (size_t)l * DMODEL * DMODEL, DMODEL, DMODEL); }
    { int n = DMODEL * 2 * FFD;
      cvt_t_kernel<<<(n + 255) / 256, 256, 0, stream>>>(
        w_ff1 + (size_t)l * DMODEL * 2 * FFD, WF1 + (size_t)l * 2 * FFD * DMODEL, DMODEL, 2 * FFD); }
    { int n = FFD * DMODEL;
      cvt_t_kernel<<<(n + 255) / 256, 256, 0, stream>>>(
        w_ff2 + (size_t)l * FFD * DMODEL, WF2 + (size_t)l * DMODEL * FFD, FFD, DMODEL); }
  }

  // concat modalities
  concat_kernel<<<(BN * DMODEL) / 256, 256, 0, stream>>>(x, x_cont, X);

  for (int l = 0; l < NLAYER; ++l) {
    // h = LN1(x)
    ln_kernel<<<BN / 4, 128, 0, stream>>>(X, ln1_g + l * DMODEL, ln1_b + l * DMODEL, HRES, H16);
    // qkv = h @ Wqkv  -> f16 [BN][768]
    gemm16<<<dim3(BN / 128, 768 / 64), 128, 0, stream>>>(
        H16, DMODEL, WQ + (size_t)l * 768 * DMODEL, DMODEL, DMODEL,
        nullptr, BIG, 768, nullptr, nullptr);
    // attention with ips masking
    attn_kernel<<<dim3(BATCH, NHEAD), 64, ATT_SMEM, stream>>>(
        BIG, cat_mask, con_mask, cat_ips, con_ips, O16);
    // x = o @ Wout + b_out + h
    gemm16<<<dim3(BN / 128, DMODEL / 64), 128, 0, stream>>>(
        O16, DMODEL, WO + (size_t)l * DMODEL * DMODEL, DMODEL, DMODEL,
        X, nullptr, DMODEL, b_out + l * DMODEL, HRES);
    // h2 = LN2(x)
    ln_kernel<<<BN / 4, 128, 0, stream>>>(X, ln2_g + l * DMODEL, ln2_b + l * DMODEL, HRES, H16);
    // ffin = a * gelu(g), fused with FF1 GEMM -> f16 [BN][1024]
    ff1_geglu_kernel<<<dim3(BN / 64, FFD / 64), 128, 0, stream>>>(
        H16, WF1 + (size_t)l * 2 * FFD * DMODEL, b_ff1 + l * 2 * FFD, BIG);
    // x = ffin @ Wff2 + b_ff2 + h2   (last layer writes d_out)
    gemm16<<<dim3(BN / 128, DMODEL / 64), 128, 0, stream>>>(
        BIG, FFD, WF2 + (size_t)l * DMODEL * FFD, FFD, FFD,
        (l == NLAYER - 1) ? out : X, nullptr, DMODEL, b_ff2 + l * DMODEL, HRES);
  }
}
